// Self_Attention2D_44032004718620
// MI455X (gfx1250) — matrix operand order repaired and checked by reading, not compiled, not run on hardware
//
#include <hip/hip_runtime.h>

// ---------------------------------------------------------------------------
// CDNA5 (gfx1250) self-attention 2D, bf16 WMMA + async-LDS.
//   B=4, H=W=64 -> N=4096, C=256, d=C/4=64
//   d_out = [ out (4*4096*256 f32) | beta (4*4096*4096 f32) ]
//   d_ws  = [ f bf16 16384x64 | g bf16 16384x64 | h bf16 16384x256 ]  (12 MB)
// ---------------------------------------------------------------------------

typedef __attribute__((ext_vector_type(16))) __bf16 v16bf;
typedef __attribute__((ext_vector_type(8)))  __bf16 v8bf;
typedef __attribute__((ext_vector_type(8)))  float  v8f;
typedef __attribute__((ext_vector_type(4)))  unsigned u32x4;

#define BATCH 4
#define NPIX  4096      // H*W
#define CCH   256
#define DQK   64
#define MROWS 16384     // BATCH*NPIX
#define NKT   (NPIX / 16)

__device__ __forceinline__ v8f wmma_bf16(const v16bf& a, const v16bf& b, v8f c) {
  // v_wmma_f32_16x16x32_bf16: D = A(16x32) * B(32x16) + C(16x16 f32)
  return __builtin_amdgcn_wmma_f32_16x16x32_bf16(
      /*neg_a=*/false, a, /*neg_b=*/false, b,
      /*c_mod=*/(short)0, c, /*reuse_a=*/false, /*reuse_b=*/false);
}

// A fragment (16x32 bf16, M x K) from row-major LDS tile, leading dim `ld`.
// Lane l holds row M = l&15; lane-half selects K sub-octet per ISA layout.
// Each half of the 16 elements is 8 contiguous bf16 -> 2x ds_load_b128.
__device__ __forceinline__ v16bf load_A_frag(const __bf16* lds, int ld, int koff,
                                             int lane) {
  const int m    = lane & 15;
  const int base = (lane >> 4) * 8;
  v16bf a;
#pragma unroll
  for (int j = 0; j < 16; ++j) {
    const int k = koff + (j >> 3) * 16 + base + (j & 7);
    a[j] = lds[m * ld + k];
  }
  return a;
}

// B fragment (32x16 bf16, K x N) from row-major LDS tile: lane&15 = N column,
// element j = K row (same K order as load_A_frag).
__device__ __forceinline__ v16bf load_B_frag(const __bf16* lds, int ld, int lane) {
  v16bf b;
#pragma unroll
  for (int j = 0; j < 16; ++j)
    b[j] = lds[((j >> 3) * 16 + (lane >> 4) * 8 + (j & 7)) * ld + (lane & 15)];
  return b;
}

// ---------------------------------------------------------------------------
// Kernel 1: projections  f = x*Wf, g = x*Wg (N-dim 64), h = x*Wh (N-dim 256)
// One wave computes one 16x16 output tile; K = 256 in steps of 32.
// grid = (MROWS/16, 4 + 4 + 16), block = 32
// ---------------------------------------------------------------------------
__global__ void proj_kernel(const float* __restrict__ x,
                            const float* __restrict__ Wf,
                            const float* __restrict__ Wg,
                            const float* __restrict__ Wh,
                            __bf16* __restrict__ fout,
                            __bf16* __restrict__ gout,
                            __bf16* __restrict__ hout) {
  __shared__ __bf16 ldsA[16 * 32];
  __shared__ __bf16 ldsB[32 * 16];
  const int lane = threadIdx.x;
  const int m0   = blockIdx.x * 16;
  const int yt   = blockIdx.y;

  const float* W; __bf16* out; int n0, ldn;
  if (yt < 4)       { W = Wf; out = fout; n0 = yt * 16;       ldn = DQK; }
  else if (yt < 8)  { W = Wg; out = gout; n0 = (yt - 4) * 16; ldn = DQK; }
  else              { W = Wh; out = hout; n0 = (yt - 8) * 16; ldn = CCH; }

  v8f acc = {};
  for (int k0 = 0; k0 < CCH; k0 += 32) {
    {  // stage A: x[m0..+15, k0..+31]  fp32 -> bf16
      const int row = lane & 15, half = lane >> 4;
      const float* src = x + (size_t)(m0 + row) * CCH + k0 + half * 16;
      __bf16* dst = ldsA + row * 32 + half * 16;
#pragma unroll
      for (int j = 0; j < 16; ++j) dst[j] = (__bf16)src[j];
    }
    {  // stage B: W[k0+lane, n0..+15]  fp32 -> bf16
      const float* src = W + (size_t)(k0 + lane) * ldn + n0;
      __bf16* dst = ldsB + lane * 16;
#pragma unroll
      for (int j = 0; j < 16; ++j) dst[j] = (__bf16)src[j];
    }
    __syncthreads();
    const v16bf a = load_A_frag(ldsA, 32, 0, lane);
    const v16bf b = load_B_frag(ldsB, 16, lane);
    acc = wmma_bf16(a, b, acc);
    __syncthreads();
  }
  const int n = lane & 15, mb = (lane >> 4) * 8;
#pragma unroll
  for (int r = 0; r < 8; ++r)
    out[(size_t)(m0 + mb + r) * ldn + n0 + n] = (__bf16)acc[r];
}

// ---------------------------------------------------------------------------
// Kernel 2: scores + two-pass streaming softmax, writes full beta (f32).
// f key tiles are streamed global->LDS with GLOBAL_LOAD_ASYNC_TO_LDS_B128
// into a ping-pong double buffer (ASYNCcnt waits, no barriers: 1 wave/WG).
// The f^T B-fragments come straight out of LDS via plain DS_LOAD_B128
// (row-major f rows are B columns, so no transpose is needed).
// grid = (NPIX/16, BATCH), block = 32
// ---------------------------------------------------------------------------

// Issue one 16x64 bf16 f-tile (2 KB) into LDS: 4 async b128 per lane.
__device__ __forceinline__ void stage_f_async(const __bf16* __restrict__ fbase,
                                              int kt, unsigned bufOff, int lane) {
  const int row = lane & 15, half = lane >> 4;
  const __bf16* src = fbase + ((size_t)kt * 16 + row) * DQK + half * 32;
  const unsigned dst = bufOff + (unsigned)(row * DQK + half * 32) * 2u;
  asm volatile(
      "global_load_async_to_lds_b128 %0, %1, off\n\t"
      "global_load_async_to_lds_b128 %0, %1, off offset:16\n\t"
      "global_load_async_to_lds_b128 %0, %1, off offset:32\n\t"
      "global_load_async_to_lds_b128 %0, %1, off offset:48"
      :: "v"(dst), "v"(src) : "memory");
}

union BFrag { u32x4 u[2]; v16bf v; };

// s-tile = A(g,16x64) * f^T via 4x ds_load_b128 + 2x wmma.
__device__ __forceinline__ v8f score_tile(unsigned fTileOff, const v16bf& a0,
                                          const v16bf& a1, int lane) {
  // ds_load_b128 addressing: lane -> (row = l&15)*128B + (half = l>>4)*16B
  const unsigned addr =
      fTileOff + (unsigned)((lane & 15) * (DQK * 2) + (lane >> 4) * 16);
  BFrag b0, b1;
  asm volatile(
      "ds_load_b128 %0, %4\n\t"
      "ds_load_b128 %1, %4 offset:32\n\t"
      "ds_load_b128 %2, %4 offset:64\n\t"
      "ds_load_b128 %3, %4 offset:96\n\t"
      "s_wait_dscnt 0x0"
      : "=v"(b0.u[0]), "=v"(b0.u[1]), "=v"(b1.u[0]), "=v"(b1.u[1])
      : "v"(addr)
      : "memory");
  v8f c = {};
  c = wmma_bf16(a0, b0.v, c);
  c = wmma_bf16(a1, b1.v, c);
  return c;
}

__global__ void attn_softmax_kernel(const __bf16* __restrict__ g,
                                    const __bf16* __restrict__ f,
                                    float* __restrict__ beta) {
  __shared__ __bf16 ldsG[16 * DQK];
  __shared__ __bf16 ldsF[2][16 * DQK];
  const int lane = threadIdx.x;
  const int m0   = blockIdx.x * 16;
  const int b    = blockIdx.y;
  const __bf16* gbase = g + (size_t)b * NPIX * DQK;
  const __bf16* fbase = f + (size_t)b * NPIX * DQK;
  const unsigned fOff0 = (unsigned)(size_t)&ldsF[0][0];
  const unsigned fOff1 = (unsigned)(size_t)&ldsF[1][0];

  {  // stage g block [16 x 64] once; A fragments live in VGPRs all kernel
    const int row = lane & 15, half = lane >> 4;
    const v8bf* src =
        (const v8bf*)(gbase + (size_t)(m0 + row) * DQK + half * 32);
    v8bf* dst = (v8bf*)(ldsG + row * DQK + half * 32);
#pragma unroll
    for (int j = 0; j < 4; ++j) dst[j] = src[j];
  }
  __syncthreads();
  const v16bf a0 = load_A_frag(ldsG, DQK, 0,  lane);
  const v16bf a1 = load_A_frag(ldsG, DQK, 32, lane);
  __syncthreads();

  float rmax[8], rsum[8];
#pragma unroll
  for (int r = 0; r < 8; ++r) { rmax[r] = -3.0e38f; rsum[r] = 0.0f; }

  // ---- pass 1: streaming max / sum-of-exp (flash-style stats) ----
  stage_f_async(fbase, 0, fOff0, lane);
  for (int kt = 0; kt < NKT; ++kt) {
    const unsigned cur = (kt & 1) ? fOff1 : fOff0;
    const unsigned nxt = (kt & 1) ? fOff0 : fOff1;
    if (kt + 1 < NKT) {
      stage_f_async(fbase, kt + 1, nxt, lane);
      asm volatile("s_wait_asynccnt 0x4" ::: "memory");  // current tile done
    } else {
      asm volatile("s_wait_asynccnt 0x0" ::: "memory");
    }
    v8f c = score_tile(cur, a0, a1, lane);
#pragma unroll
    for (int r = 0; r < 8; ++r) {
      float v = c[r];
      v = fmaxf(v, __shfl_xor(v, 1));
      v = fmaxf(v, __shfl_xor(v, 2));
      v = fmaxf(v, __shfl_xor(v, 4));
      v = fmaxf(v, __shfl_xor(v, 8));   // row max within 16-lane half
      const float mnew = fmaxf(rmax[r], v);
      float se = __expf(c[r] - mnew);
      se += __shfl_xor(se, 1);
      se += __shfl_xor(se, 2);
      se += __shfl_xor(se, 4);
      se += __shfl_xor(se, 8);
      rsum[r] = rsum[r] * __expf(rmax[r] - mnew) + se;
      rmax[r] = mnew;
    }
  }
  float rinv[8];
#pragma unroll
  for (int r = 0; r < 8; ++r) rinv[r] = 1.0f / rsum[r];

  // ---- pass 2: recompute scores, write normalized beta (256 MB output) ----
  float* brow = beta + (size_t)(b * NPIX + m0) * NPIX;
  const int n = lane & 15, mb = (lane >> 4) * 8;
  stage_f_async(fbase, 0, fOff0, lane);
  for (int kt = 0; kt < NKT; ++kt) {
    const unsigned cur = (kt & 1) ? fOff1 : fOff0;
    const unsigned nxt = (kt & 1) ? fOff0 : fOff1;
    if (kt + 1 < NKT) {
      stage_f_async(fbase, kt + 1, nxt, lane);
      asm volatile("s_wait_asynccnt 0x4" ::: "memory");
    } else {
      asm volatile("s_wait_asynccnt 0x0" ::: "memory");
    }
    v8f c = score_tile(cur, a0, a1, lane);
#pragma unroll
    for (int r = 0; r < 8; ++r)
      brow[(size_t)(mb + r) * NPIX + kt * 16 + n] =
          __expf(c[r] - rmax[r]) * rinv[r];
  }
}

// ---------------------------------------------------------------------------
// Kernel 3: o = beta @ h ; out = gamma*o + x
// h tiles streamed global->LDS via async DMA, overlapped with the beta
// fp32->bf16 conversion (which must pass through VGPRs anyway).
// One wave computes a 16(M) x 64(N) tile; K = 4096 in steps of 32.
// grid = (CCH/64, NPIX/16, BATCH), block = 32
// ---------------------------------------------------------------------------
__global__ void attn_out_kernel(const float* __restrict__ beta,
                                const __bf16* __restrict__ h,
                                const float* __restrict__ x,
                                const float* __restrict__ gamma,
                                float* __restrict__ out) {
  __shared__ __bf16 ldsA[16 * 32];
  __shared__ __bf16 ldsB[32 * 64];
  const int lane = threadIdx.x;
  const int n0   = blockIdx.x * 64;
  const int m0   = blockIdx.y * 16;
  const int b    = blockIdx.z;
  const float*  brow  = beta + (size_t)(b * NPIX + m0) * NPIX;
  const __bf16* hbase = h + (size_t)b * NPIX * CCH;
  const unsigned ldsBLane = (unsigned)(size_t)&ldsB[0] + (unsigned)lane * 128u;

  v8f acc[4] = {};
  for (int k0 = 0; k0 < NPIX; k0 += 32) {
    {  // async: h tile [32 keys x 64 cols], one key row (128B) per lane
      const __bf16* src = hbase + (size_t)(k0 + lane) * CCH + n0;
      asm volatile(
          "global_load_async_to_lds_b128 %0, %1, off\n\t"
          "global_load_async_to_lds_b128 %0, %1, off offset:16\n\t"
          "global_load_async_to_lds_b128 %0, %1, off offset:32\n\t"
          "global_load_async_to_lds_b128 %0, %1, off offset:48\n\t"
          "global_load_async_to_lds_b128 %0, %1, off offset:64\n\t"
          "global_load_async_to_lds_b128 %0, %1, off offset:80\n\t"
          "global_load_async_to_lds_b128 %0, %1, off offset:96\n\t"
          "global_load_async_to_lds_b128 %0, %1, off offset:112"
          :: "v"(ldsBLane), "v"(src) : "memory");
    }
    {  // stage beta tile [16 x 32] fp32 -> bf16 (overlaps the async DMA)
      const int row = lane & 15, half = lane >> 4;
      const float* src = brow + (size_t)row * NPIX + k0 + half * 16;
      __bf16* dst = ldsA + row * 32 + half * 16;
#pragma unroll
      for (int j = 0; j < 16; ++j) dst[j] = (__bf16)src[j];
    }
    __syncthreads();
    asm volatile("s_wait_asynccnt 0x0" ::: "memory");
    const v16bf a = load_A_frag(ldsA, 32, 0, lane);
#pragma unroll
    for (int q = 0; q < 4; ++q) {
      const v16bf bq = load_B_frag(ldsB + q * 16, 64, lane);
      acc[q] = wmma_bf16(a, bq, acc[q]);
    }
    __syncthreads();
  }
  const float gm = gamma[0];
  const int n = lane & 15, mb = (lane >> 4) * 8;
#pragma unroll
  for (int q = 0; q < 4; ++q)
#pragma unroll
    for (int r = 0; r < 8; ++r) {
      const size_t idx =
          (size_t)(b * NPIX + m0 + mb + r) * CCH + n0 + q * 16 + n;
      out[idx] = gm * acc[q][r] + x[idx];
    }
}

// ---------------------------------------------------------------------------
extern "C" void kernel_launch(void* const* d_in, const int* in_sizes, int n_in,
                              void* d_out, int out_size, void* d_ws,
                              size_t ws_size, hipStream_t stream) {
  const float* x     = (const float*)d_in[0];   // [4,64,64,256]
  const float* Wf    = (const float*)d_in[1];   // [256,64]
  const float* Wg    = (const float*)d_in[2];   // [256,64]
  const float* Wh    = (const float*)d_in[3];   // [256,256]
  const float* gamma = (const float*)d_in[4];   // [1]

  float* out  = (float*)d_out;                               // 4*4096*256
  float* beta = (float*)d_out + (size_t)BATCH * NPIX * CCH;  // 4*4096*4096

  __bf16* fw = (__bf16*)d_ws;                 // 16384*64  bf16 (2 MB)
  __bf16* gw = fw + (size_t)MROWS * DQK;      // 16384*64  bf16 (2 MB)
  __bf16* hw = gw + (size_t)MROWS * DQK;      // 16384*256 bf16 (8 MB)

  proj_kernel<<<dim3(MROWS / 16, 24), 32, 0, stream>>>(x, Wf, Wg, Wh, fw, gw, hw);
  attn_softmax_kernel<<<dim3(NPIX / 16, BATCH), 32, 0, stream>>>(gw, fw, beta);
  attn_out_kernel<<<dim3(CCH / 64, NPIX / 16, BATCH), 32, 0, stream>>>(
      beta, hw, x, gamma, out);
}
